// PointNet2Backbone_83726092468517
// MI455X (gfx1250) — compile-verified, matrix operand order checked by reading, not run on hardware
//
#include <hip/hip_runtime.h>

typedef __attribute__((ext_vector_type(16))) _Float16 v16h;
typedef __attribute__((ext_vector_type(8)))  _Float16 v8h;
typedef __attribute__((ext_vector_type(8)))  float    v8f;

#define NBATCH 8
#define NPTS   8192

// ---------------------------------------------------------------------------
__global__ void split_kernel(const float* __restrict__ pts,
                             float* __restrict__ xyz, float* __restrict__ feat, int n) {
  int i = blockIdx.x * blockDim.x + threadIdx.x;
  if (i < n) {
    xyz[i * 3 + 0] = pts[i * 5 + 1];
    xyz[i * 3 + 1] = pts[i * 5 + 2];
    xyz[i * 3 + 2] = pts[i * 5 + 3];
    feat[i] = pts[i * 5 + 4];
  }
}

__global__ void coords_kernel(const float* __restrict__ pts, float* __restrict__ out, int n) {
  int i = blockIdx.x * blockDim.x + threadIdx.x;
  if (i < n) {
    out[i * 4 + 0] = pts[i * 5 + 0];
    out[i * 4 + 1] = pts[i * 5 + 1];
    out[i * 4 + 2] = pts[i * 5 + 2];
    out[i * 4 + 3] = pts[i * 5 + 3];
  }
}

// ---------------------------------------------------------------------------
// Pack W [K,N] f32 into CDNA5 B-fragment tile order, zero-padded to Kpad.
__global__ void pack_w_kernel(const float* __restrict__ W, int K, int N,
                              _Float16* __restrict__ Wp, int total) {
  int idx = blockIdx.x * blockDim.x + threadIdx.x;
  if (idx >= total) return;
  int e    = idx & 15;
  int lane = (idx >> 4) & 31;
  int tile = idx >> 9;
  int ntiles = N >> 4;
  int nt = tile % ntiles;
  int kt = tile / ntiles;
  int ml = lane & 15, half = lane >> 4;
  int kk = ((e < 8) ? e : e + 8) + (half << 3) + kt * 32;
  int n  = nt * 16 + ml;
  Wp[idx] = (kk < K) ? (_Float16)W[(size_t)kk * N + n] : (_Float16)0.f;
}

// ---------------------------------------------------------------------------
// Farthest point sampling: one block per batch. dists lives in LDS.
__global__ __launch_bounds__(1024)
void fps_kernel(const float* __restrict__ xyz, int N, int npoint, int* __restrict__ idx) {
  __shared__ float dists[NPTS];
  __shared__ float rv[1024];
  __shared__ int   ri[1024];
  int b = blockIdx.x, tid = threadIdx.x;
  const float* P = xyz + (size_t)b * N * 3;
  for (int i = tid; i < N; i += 1024) dists[i] = 1e10f;
  if (tid == 0) idx[b * npoint] = 0;
  __syncthreads();
  int last = 0;
  for (int s = 1; s < npoint; ++s) {
    float lx = P[last * 3 + 0], ly = P[last * 3 + 1], lz = P[last * 3 + 2];
    float best = -1.0f; int besti = 0;
    for (int i = tid; i < N; i += 1024) {
      float dx = P[i * 3 + 0] - lx;
      float dy = P[i * 3 + 1] - ly;
      float dz = P[i * 3 + 2] - lz;
      float d = dx * dx + dy * dy + dz * dz;
      float nd = fminf(dists[i], d);
      dists[i] = nd;
      if (nd > best || (nd == best && i < besti)) { best = nd; besti = i; }
    }
    rv[tid] = best; ri[tid] = besti;
    __syncthreads();
    for (int off = 512; off > 0; off >>= 1) {
      if (tid < off) {
        float ov = rv[tid + off]; int oi = ri[tid + off];
        if (ov > rv[tid] || (ov == rv[tid] && oi < ri[tid])) { rv[tid] = ov; ri[tid] = oi; }
      }
      __syncthreads();
    }
    last = ri[0];
    if (tid == 0) idx[b * npoint + s] = last;
    __syncthreads();
  }
}

__global__ void gather_xyz_kernel(const float* __restrict__ xyz, int N,
                                  const int* __restrict__ idx, int np,
                                  float* __restrict__ out, int total) {
  int t = blockIdx.x * blockDim.x + threadIdx.x;
  if (t >= total) return;
  int b = t / np, i = t - b * np;
  int id = idx[b * np + i];
  const float* src = xyz + ((size_t)b * N + id) * 3;
  out[(size_t)t * 3 + 0] = src[0];
  out[(size_t)t * 3 + 1] = src[1];
  out[(size_t)t * 3 + 2] = src[2];
}

// ---------------------------------------------------------------------------
// Ball query + grouping: one wave per query point.
__global__ __launch_bounds__(32)
void group_kernel(const float* __restrict__ xyz, int N,
                  const float* __restrict__ feats, int Cf,
                  const float* __restrict__ nx, int nx_bstride, int S, int nsample,
                  float r2, _Float16* __restrict__ A, int lda) {
  __shared__ int   slist[512];
  __shared__ float sd2[32];
  int gid = blockIdx.x;
  int b = gid / S, s = gid - b * S;
  int lane = threadIdx.x;
  const float* P = xyz + (size_t)b * N * 3;
  const float* F = feats + (size_t)b * N * Cf;
  const float* Q = nx + (size_t)b * nx_bstride + (size_t)s * 3;
  float qx = Q[0], qy = Q[1], qz = Q[2];
  int cnt = 0;
  for (int base = 0; base < N && cnt < nsample; base += 32) {
    int i = base + lane;
    float d2 = 3.0e38f;
    if (i < N) {
      float dx = P[i * 3 + 0] - qx;
      float dy = P[i * 3 + 1] - qy;
      float dz = P[i * 3 + 2] - qz;
      d2 = dx * dx + dy * dy + dz * dz;
    }
    sd2[lane] = d2;
    __syncthreads();
    for (int t = 0; t < 32; ++t) {
      if (cnt >= nsample) break;
      if (sd2[t] < r2) { if (lane == 0) slist[cnt] = base + t; cnt++; }
    }
    __syncthreads();
  }
  int empty = (cnt == 0);
  int first = 0;
  if (!empty) first = slist[0];
  size_t rowbase = (size_t)gid * nsample;
  for (int j = lane; j < nsample; j += 32) {
    int id = (j < cnt) ? slist[j] : first;
    if (empty) id = 0;
    _Float16* row = A + (rowbase + j) * lda;
    float gx = P[id * 3 + 0] - qx;
    float gy = P[id * 3 + 1] - qy;
    float gz = P[id * 3 + 2] - qz;
    if (empty) { gx = 0.f; gy = 0.f; gz = 0.f; }
    row[0] = (_Float16)gx; row[1] = (_Float16)gy; row[2] = (_Float16)gz;
    for (int c = 0; c < Cf; ++c)
      row[3 + c] = empty ? (_Float16)0.f : (_Float16)F[(size_t)id * Cf + c];
    for (int c = 3 + Cf; c < lda; ++c) row[c] = (_Float16)0.f;
  }
}

// ---------------------------------------------------------------------------
// WMMA GEMM + ReLU, no LDS. A:[M,lda] f16 (lda = Kpad, multiple of 32, rows
// 64B-aligned, pad cols zero). Wp: packed fragment tiles. One wave computes a
// 16x32 output block (two 16x16 tiles sharing the A fragment); the tile-count
// decision is hoisted out of the k-loop so the hot loop is loads + wmma only.
__global__ __launch_bounds__(32)
void wmma_gemm(const _Float16* __restrict__ A, int lda,
               const _Float16* __restrict__ Wp, int N,
               _Float16* __restrict__ out16, float* __restrict__ out32,
               int ldo, int relu) {
  int lane = threadIdx.x;
  int ml = lane & 15, half = lane >> 4;
  size_t m0 = (size_t)blockIdx.x * 16;
  int ntiles = N >> 4;
  int nt0 = blockIdx.y * 2;
  bool two = (nt0 + 1) < ntiles;

  const _Float16* arow = A + (m0 + ml) * lda + (half << 3);
  const _Float16* bp0  = Wp + ((size_t)nt0 * 32 + lane) * 16;
  const _Float16* bp1  = bp0 + 512;
  size_t bstride = (size_t)ntiles * 512;

  v8f acc0 = {}, acc1 = {};
  if (two) {
    for (int k0 = 0; k0 < lda; k0 += 32) {
      v8h alo = *(const v8h*)(arow + k0);
      v8h ahi = *(const v8h*)(arow + k0 + 16);
      v16h av;
#pragma unroll
      for (int e = 0; e < 8; ++e) { av[e] = alo[e]; av[e + 8] = ahi[e]; }
      if (k0 + 32 < lda) __builtin_prefetch(arow + k0 + 32, 0, 3);
      v16h bv0 = *(const v16h*)bp0;
      v16h bv1 = *(const v16h*)bp1;
      acc0 = __builtin_amdgcn_wmma_f32_16x16x32_f16(false, av, false, bv0,
                                                    (short)0, acc0, false, false);
      acc1 = __builtin_amdgcn_wmma_f32_16x16x32_f16(false, av, false, bv1,
                                                    (short)0, acc1, false, false);
      bp0 += bstride; bp1 += bstride;
    }
  } else {
    for (int k0 = 0; k0 < lda; k0 += 32) {
      v8h alo = *(const v8h*)(arow + k0);
      v8h ahi = *(const v8h*)(arow + k0 + 16);
      v16h av;
#pragma unroll
      for (int e = 0; e < 8; ++e) { av[e] = alo[e]; av[e + 8] = ahi[e]; }
      if (k0 + 32 < lda) __builtin_prefetch(arow + k0 + 32, 0, 3);
      v16h bv0 = *(const v16h*)bp0;
      acc0 = __builtin_amdgcn_wmma_f32_16x16x32_f16(false, av, false, bv0,
                                                    (short)0, acc0, false, false);
      bp0 += bstride;
    }
  }

  int mh = half * 8;
  auto store_tile = [&](v8f& acc, int nt) {
    int col = nt * 16 + ml;
#pragma unroll
    for (int r = 0; r < 8; ++r) {
      float v = acc[r];
      if (relu) v = v > 0.f ? v : 0.f;
      size_t row = m0 + mh + r;
      if (out16) out16[row * (size_t)ldo + col] = (_Float16)v;
      if (out32) out32[row * (size_t)ldo + col] = v;
    }
  };
  store_tile(acc0, nt0);
  if (two) store_tile(acc1, nt0 + 1);

  // zero-fill pad columns [N, ldo) so this buffer can feed the next layer
  if (out16 && ldo > N && (nt0 + (two ? 2 : 1)) == ntiles) {
    for (int c = N + ml; c < ldo; c += 16) {
#pragma unroll
      for (int r = 0; r < 8; ++r)
        out16[(m0 + mh + r) * (size_t)ldo + c] = (_Float16)0.f;
    }
  }
}

// ---------------------------------------------------------------------------
__global__ void maxpool_kernel(const _Float16* __restrict__ act, int ld, int nsample, int C,
                               float* __restrict__ out, int ldo, int coff) {
  int gid = blockIdx.x;
  for (int c = threadIdx.x; c < C; c += blockDim.x) {
    float m = -3.0e38f;
    for (int j = 0; j < nsample; ++j)
      m = fmaxf(m, (float)act[((size_t)gid * nsample + j) * ld + c]);
    out[(size_t)gid * ldo + coff + c] = m;
  }
}

// ---------------------------------------------------------------------------
// 3-NN inverse-distance interpolation + concat uf; one wave per unknown point.
__global__ __launch_bounds__(32)
void fp_interp_kernel(const float* __restrict__ ux, const float* __restrict__ kx,
                      const float* __restrict__ kf, int C,
                      const float* __restrict__ uf, int uf_stride, int Cu,
                      int Su, int Sk, _Float16* __restrict__ A, int lda) {
  __shared__ float sd[96];
  __shared__ int   si[96];
  int gid = blockIdx.x;
  int b = gid / Su, u = gid - b * Su;
  int lane = threadIdx.x;
  const float* Q = ux + ((size_t)b * Su + u) * 3;
  const float* P = kx + (size_t)b * Sk * 3;
  float qx = Q[0], qy = Q[1], qz = Q[2];
  float bd0 = 3e38f, bd1 = 3e38f, bd2 = 3e38f;
  int   bi0 = 0, bi1 = 0, bi2 = 0;
  for (int i = lane; i < Sk; i += 32) {
    float dx = P[i * 3 + 0] - qx;
    float dy = P[i * 3 + 1] - qy;
    float dz = P[i * 3 + 2] - qz;
    float d = dx * dx + dy * dy + dz * dz;
    if (d < bd0)      { bd2 = bd1; bi2 = bi1; bd1 = bd0; bi1 = bi0; bd0 = d; bi0 = i; }
    else if (d < bd1) { bd2 = bd1; bi2 = bi1; bd1 = d; bi1 = i; }
    else if (d < bd2) { bd2 = d; bi2 = i; }
  }
  sd[lane * 3 + 0] = bd0; si[lane * 3 + 0] = bi0;
  sd[lane * 3 + 1] = bd1; si[lane * 3 + 1] = bi1;
  sd[lane * 3 + 2] = bd2; si[lane * 3 + 2] = bi2;
  __syncthreads();
  float gd0 = 3e38f, gd1 = 3e38f, gd2 = 3e38f;
  int   gi0 = 0, gi1 = 0, gi2 = 0;
  for (int t = 0; t < 96; ++t) {
    float d = sd[t]; int ii = si[t];
    if (d < gd0)      { gd2 = gd1; gi2 = gi1; gd1 = gd0; gi1 = gi0; gd0 = d; gi0 = ii; }
    else if (d < gd1) { gd2 = gd1; gi2 = gi1; gd1 = d; gi1 = ii; }
    else if (d < gd2) { gd2 = d; gi2 = ii; }
  }
  float w0 = 1.0f / (gd0 + 1e-8f);
  float w1 = 1.0f / (gd1 + 1e-8f);
  float w2 = 1.0f / (gd2 + 1e-8f);
  float wsum = w0 + w1 + w2;
  w0 /= wsum; w1 /= wsum; w2 /= wsum;
  const float* KF = kf + (size_t)b * Sk * C;
  _Float16* row = A + (size_t)gid * lda;
  for (int c = lane; c < C; c += 32) {
    float v = KF[(size_t)gi0 * C + c] * w0 +
              KF[(size_t)gi1 * C + c] * w1 +
              KF[(size_t)gi2 * C + c] * w2;
    row[c] = (_Float16)v;
  }
  const float* UF = uf + (size_t)gid * uf_stride;
  for (int c = lane; c < Cu; c += 32) row[C + c] = (_Float16)UF[c];
  for (int c = lane; c < lda; c += 32)
    if (c >= C + Cu) row[c] = (_Float16)0.f;
}

// ===========================================================================
extern "C" void kernel_launch(void* const* d_in, const int* in_sizes, int n_in,
                              void* d_out, int out_size, void* d_ws, size_t ws_size,
                              hipStream_t stream) {
  (void)in_sizes; (void)n_in; (void)out_size; (void)ws_size;
  const float* points = (const float*)d_in[0];

  // weights (dict order: sa1(6), sa2(6), fp1(2), fp2(2), glob(2))
  static const int wK[18] = {4, 16, 16, 4, 32, 32,
                             99, 64, 64, 99, 64, 96,
                             260, 128, 352, 256, 259, 256};
  static const int wN[18] = {16, 16, 32, 32, 32, 64,
                             64, 64, 128, 64, 96, 128,
                             128, 128, 256, 256, 256, 512};
  const float* wsrc[18];
  for (int i = 0; i < 18; ++i) wsrc[i] = (const float*)d_in[1 + i];

  // ---- workspace carve-up
  char* ws = (char*)d_ws;
  size_t o = 0;
  auto carve = [&](size_t bytes) { size_t r = o; o += (bytes + 255) & ~(size_t)255; return r; };
  float* xyz   = (float*)(ws + carve((size_t)NBATCH * NPTS * 3 * 4));
  float* feat  = (float*)(ws + carve((size_t)NBATCH * NPTS * 4));
  int*   idx1  = (int*)  (ws + carve((size_t)NBATCH * 2048 * 4));
  float* nx1   = (float*)(ws + carve((size_t)NBATCH * 2048 * 3 * 4));
  float* f1    = (float*)(ws + carve((size_t)NBATCH * 2048 * 96 * 4));
  int*   idx2  = (int*)  (ws + carve((size_t)NBATCH * 512 * 4));
  float* nx2   = (float*)(ws + carve((size_t)NBATCH * 512 * 3 * 4));
  float* f2    = (float*)(ws + carve((size_t)NBATCH * 512 * 256 * 4));
  float* f1up  = (float*)(ws + carve((size_t)NBATCH * 2048 * 256 * 4));
  _Float16* wpool = (_Float16*)(ws + carve((size_t)1 << 20));
  _Float16* buf0  = (_Float16*)(ws + carve((size_t)40 << 20));
  _Float16* buf1  = (_Float16*)(ws + carve((size_t)64 << 20));

  // d_out: point_features [65536,128] | point_coords [65536,4] | gfeat [8,512]
  float* out_pf = (float*)d_out;
  float* out_pc = out_pf + (size_t)NBATCH * NPTS * 128;
  float* out_g  = out_pc + (size_t)NBATCH * NPTS * 4;

  // ---- pack weights into fragment tiles (zero-padded to Kpad)
  _Float16* wp[18];
  {
    size_t acc = 0;
    for (int i = 0; i < 18; ++i) {
      int Kpad = (wK[i] + 31) & ~31;
      int total = (Kpad >> 5) * (wN[i] >> 4) * 512;
      wp[i] = wpool + acc;
      acc += (size_t)total;
      pack_w_kernel<<<(total + 255) / 256, 256, 0, stream>>>(wsrc[i], wK[i], wN[i], wp[i], total);
    }
  }

  // ---- split points / coords
  {
    int n = NBATCH * NPTS;
    split_kernel<<<(n + 255) / 256, 256, 0, stream>>>(points, xyz, feat, n);
    coords_kernel<<<(n + 255) / 256, 256, 0, stream>>>(points, out_pc, n);
  }

  auto gemm = [&](const _Float16* A, int lda, int wi,
                  _Float16* o16, float* o32, int ldo, int M) {
    int ntiles = wN[wi] >> 4;
    dim3 grid(M / 16, (ntiles + 1) / 2);
    wmma_gemm<<<grid, 32, 0, stream>>>(A, lda, wp[wi], wN[wi], o16, o32, ldo, 1);
  };

  // ---- FPS level 1 + gather
  fps_kernel<<<NBATCH, 1024, 0, stream>>>(xyz, NPTS, 2048, idx1);
  {
    int tot = NBATCH * 2048;
    gather_xyz_kernel<<<(tot + 255) / 256, 256, 0, stream>>>(xyz, NPTS, idx1, 2048, nx1, tot);
  }

  // ---- SA1 branch 1: r2=1, ns=16, MLP 4->16->16->32 -> f1[:, 0:32]
  {
    group_kernel<<<NBATCH * 2048, 32, 0, stream>>>(xyz, NPTS, feat, 1, nx1, 2048 * 3,
                                                   2048, 16, 1.0f, buf0, 32);
    int M = NBATCH * 2048 * 16;
    gemm(buf0, 32, 0, buf1, nullptr, 32, M);   // N=16, pad to 32
    gemm(buf1, 32, 1, buf0, nullptr, 32, M);   // N=16, pad to 32
    gemm(buf0, 32, 2, buf1, nullptr, 32, M);   // N=32
    maxpool_kernel<<<NBATCH * 2048, 256, 0, stream>>>(buf1, 32, 16, 32, f1, 96, 0);
  }
  // ---- SA1 branch 2: r2=4, ns=32, MLP 4->32->32->64 -> f1[:, 32:96]
  {
    group_kernel<<<NBATCH * 2048, 32, 0, stream>>>(xyz, NPTS, feat, 1, nx1, 2048 * 3,
                                                   2048, 32, 4.0f, buf0, 32);
    int M = NBATCH * 2048 * 32;
    gemm(buf0, 32, 3, buf1, nullptr, 32, M);
    gemm(buf1, 32, 4, buf0, nullptr, 32, M);
    gemm(buf0, 32, 5, buf1, nullptr, 64, M);
    maxpool_kernel<<<NBATCH * 2048, 256, 0, stream>>>(buf1, 64, 32, 64, f1, 96, 32);
  }

  // ---- FPS level 2 + gather
  fps_kernel<<<NBATCH, 1024, 0, stream>>>(nx1, 2048, 512, idx2);
  {
    int tot = NBATCH * 512;
    gather_xyz_kernel<<<(tot + 255) / 256, 256, 0, stream>>>(nx1, 2048, idx2, 512, nx2, tot);
  }

  // ---- SA2 branch 1: r2=4, ns=16, MLP 99->64->64->128 -> f2[:, 0:128]
  {
    group_kernel<<<NBATCH * 512, 32, 0, stream>>>(nx1, 2048, f1, 96, nx2, 512 * 3,
                                                  512, 16, 4.0f, buf0, 128);
    int M = NBATCH * 512 * 16;
    gemm(buf0, 128, 6, buf1, nullptr, 64, M);
    gemm(buf1, 64, 7, buf0, nullptr, 64, M);
    gemm(buf0, 64, 8, buf1, nullptr, 128, M);
    maxpool_kernel<<<NBATCH * 512, 256, 0, stream>>>(buf1, 128, 16, 128, f2, 256, 0);
  }
  // ---- SA2 branch 2: r2=16, ns=32, MLP 99->64->96->128 -> f2[:, 128:256]
  {
    group_kernel<<<NBATCH * 512, 32, 0, stream>>>(nx1, 2048, f1, 96, nx2, 512 * 3,
                                                  512, 32, 16.0f, buf0, 128);
    int M = NBATCH * 512 * 32;
    gemm(buf0, 128, 9, buf1, nullptr, 64, M);
    gemm(buf1, 64, 10, buf0, nullptr, 96, M);
    gemm(buf0, 96, 11, buf1, nullptr, 128, M);
    maxpool_kernel<<<NBATCH * 512, 256, 0, stream>>>(buf1, 128, 32, 128, f2, 256, 128);
  }

  // ---- FP2: interp(new_xyz1 <- new_xyz2, f2) ++ f1 -> MLP 352->256->256 -> f1up
  {
    fp_interp_kernel<<<NBATCH * 2048, 32, 0, stream>>>(nx1, nx2, f2, 256, f1, 96, 96,
                                                       2048, 512, buf0, 352);
    int M = NBATCH * 2048;
    gemm(buf0, 352, 14, buf1, nullptr, 256, M);
    gemm(buf1, 256, 15, nullptr, f1up, 256, M);
  }

  // ---- FP1: interp(xyz <- new_xyz1, f1up) ++ feat0 -> MLP 260->128->128 -> d_out
  {
    fp_interp_kernel<<<NBATCH * NPTS, 32, 0, stream>>>(xyz, nx1, f1up, 256,
                                                       points + 1, 5, 4,
                                                       NPTS, 2048, buf0, 288);
    int M = NBATCH * NPTS;
    gemm(buf0, 288, 12, buf1, nullptr, 128, M);
    gemm(buf1, 128, 13, nullptr, out_pf, 128, M);
  }

  // ---- Global SA: all 512 points, MLP 259->256->512, max -> gfeat [8,512]
  {
    group_kernel<<<NBATCH, 32, 0, stream>>>(nx2, 512, f2, 256, nx2, 512 * 3,
                                            1, 512, 1.0e12f, buf0, 288);
    int M = NBATCH * 512;
    gemm(buf0, 288, 16, buf1, nullptr, 256, M);
    gemm(buf1, 256, 17, buf0, nullptr, 512, M);
    maxpool_kernel<<<NBATCH, 256, 0, stream>>>(buf0, 512, 512, 512, out_g, 512, 0);
  }
}
